// GraphormerNodeEncoder_62122406969754
// MI455X (gfx1250) — compile-verified, matrix operand order checked by reading, not run on hardware
//
#include <hip/hip_runtime.h>
#include <hip/hip_bf16.h>
#include <math.h>

#define B_ 4
#define N_ 1024
#define D_ 256
#define H_ 8
#define L_ 4
#define FF_ 1024
#define HD_ 32
#define MAXDEG 200
#define MAXSPD 512
#define MAXPATH 5

typedef __attribute__((ext_vector_type(16))) __bf16 v16bf;
typedef __attribute__((ext_vector_type(8)))  __bf16 v8bf;
typedef __attribute__((ext_vector_type(8)))  float  v8f;

// Branchless GELU: Abramowitz & Stegun 7.1.26 erf (|err|<1.5e-7).
__device__ __forceinline__ float gelu_exact(float x) {
  float xs = x * 0.70710678118654752f;
  float ax = fabsf(xs);
  float t  = __builtin_amdgcn_rcpf(1.0f + 0.3275911f * ax);
  float y  = t * (0.254829592f + t * (-0.284496736f + t * (1.421413741f +
             t * (-1.453152027f + t * 1.061405429f))));
  float e  = __expf(-ax * ax);
  float er = copysignf(1.0f - y * e, xs);
  return 0.5f * x * (1.0f + er);
}

// Load a 16-element bf16 fragment half-pair: elems[0..7] = p[0..7],
// elems[8..15] = p[16..23]  (two 16-byte global_load_b128).
__device__ __forceinline__ v16bf ldfrag(const __bf16* __restrict__ p) {
  v8bf lo = *(const v8bf*)p;
  v8bf hi = *(const v8bf*)(p + 16);
  v16bf r;
#pragma unroll
  for (int j = 0; j < 8; ++j) { r[j] = lo[j]; r[8 + j] = hi[j]; }
  return r;
}

// ---------------------------------------------------------------------------
// f32 -> bf16 elementwise convert (4 per thread, vectorized).
// ---------------------------------------------------------------------------
__global__ __launch_bounds__(256) void conv_bf16(
    const float* __restrict__ in, __bf16* __restrict__ out, int n)
{
  int i = (blockIdx.x * 256 + threadIdx.x) * 4;
  if (i >= n) return;
  float4 xv = *(const float4*)(in + i);
  out[i]     = (__bf16)xv.x;
  out[i + 1] = (__bf16)xv.y;
  out[i + 2] = (__bf16)xv.z;
  out[i + 3] = (__bf16)xv.w;
}

// ---------------------------------------------------------------------------
// W (K x Nc, f32) -> WT (Nc x K, bf16) via LDS 32x32 tile transpose.
// grid: (Nc/32, K/32), block: 256
// ---------------------------------------------------------------------------
__global__ __launch_bounds__(256) void transpose_conv_bf16(
    const float* __restrict__ in, __bf16* __restrict__ out, int K, int Nc)
{
  __shared__ float tile[32][33];
  const int nb = blockIdx.x * 32;
  const int kb = blockIdx.y * 32;
  const int tx = threadIdx.x & 31;
  const int ty = threadIdx.x >> 5;
#pragma unroll
  for (int i = 0; i < 4; ++i)
    tile[ty + i * 8][tx] = in[(size_t)(kb + ty + i * 8) * Nc + nb + tx];
  __syncthreads();
#pragma unroll
  for (int i = 0; i < 4; ++i)
    out[(size_t)(nb + ty + i * 8) * K + kb + tx] = (__bf16)tile[tx][ty + i * 8];
}

// ---------------------------------------------------------------------------
// V (B,N,D) bf16 -> VT (B,D,N) bf16 via LDS tile transpose.
// grid: (D/32, N/32, B), block: 256
// ---------------------------------------------------------------------------
__global__ __launch_bounds__(256) void transpose_v_bf16(
    const __bf16* __restrict__ in, __bf16* __restrict__ out)
{
  __shared__ __bf16 tile[32][33];
  const int db = blockIdx.x * 32;
  const int nb = blockIdx.y * 32;
  const int b  = blockIdx.z;
  const int tx = threadIdx.x & 31;
  const int ty = threadIdx.x >> 5;
  const __bf16* ib = in + (size_t)b * N_ * D_;
  __bf16* ob = out + (size_t)b * N_ * D_;
#pragma unroll
  for (int i = 0; i < 4; ++i)
    tile[ty + i * 8][tx] = ib[(size_t)(nb + ty + i * 8) * D_ + db + tx];
  __syncthreads();
#pragma unroll
  for (int i = 0; i < 4; ++i)
    ob[(size_t)(db + ty + i * 8) * N_ + nb + tx] = tile[tx][ty + i * 8];
}

// ---------------------------------------------------------------------------
// GEMM: Y[M,Nc] = act((X @ W + bias) * scale) with X bf16 (M x K) and
// WT bf16 (Nc x K, pre-transposed). One wave computes a 16x64 strip:
// A-fragment reused across 4 v_wmma_f32_16x16x32_bf16; all operand loads
// are contiguous 16-byte b128s. Output f32 (Yf) or bf16 (Yb).
// Fragment layout per CDNA5 ISA 7.12.2 (A rows / B cols striped per lane,
// K split 0..7/16..23 (lanes 0-15) and 8..15/24..31 (lanes 16-31)).
// ---------------------------------------------------------------------------
__global__ __launch_bounds__(32) void gemm_bf16_wmma(
    const __bf16* __restrict__ X, const __bf16* __restrict__ WT,
    const float* __restrict__ bias, float* __restrict__ Yf,
    __bf16* __restrict__ Yb, int M, int K, int Nc, float scale, int act)
{
  const int lane = threadIdx.x;
  const int hl  = lane >> 4;
  const int l16 = lane & 15;
  const int tn  = blockIdx.x << 6;   // 64-wide strip
  const int tm  = blockIdx.y << 4;

  v8f acc[4];
#pragma unroll
  for (int nb = 0; nb < 4; ++nb) acc[nb] = (v8f){0.f,0.f,0.f,0.f,0.f,0.f,0.f,0.f};

  const __bf16* xrow = X + (size_t)(tm + l16) * K + hl * 8;
  const __bf16* wrow0 = WT + (size_t)(tn + l16) * K + hl * 8;

  for (int kb = 0; kb < K; kb += 32) {
    __builtin_prefetch(xrow + kb + 64, 0, 1);      // global_prefetch_b8
    v16bf a = ldfrag(xrow + kb);
#pragma unroll
    for (int nb = 0; nb < 4; ++nb) {
      v16bf b = ldfrag(wrow0 + (size_t)(nb * 16) * K + kb);
      acc[nb] = __builtin_amdgcn_wmma_f32_16x16x32_bf16(
          false, a, false, b, (short)0, acc[nb], false, false);
    }
  }

#pragma unroll
  for (int nb = 0; nb < 4; ++nb) {
    const int col = tn + nb * 16 + l16;
    const float bb = bias ? bias[col] : 0.f;
#pragma unroll
    for (int j = 0; j < 8; ++j) {
      int row = tm + j + hl * 8;
      float y = (acc[nb][j] + bb) * scale;
      if (act) y = gelu_exact(y);
      if (Yb) Yb[(size_t)row * Nc + col] = (__bf16)y;
      else    Yf[(size_t)row * Nc + col] = y;
    }
  }
}

// ---------------------------------------------------------------------------
// attn_bias[b,h,x,y] = spatial_tab[min(dist,512)+1, h]
//                    + (sum_l path_data[b,x,y,l] * path_tab[l*H+h]) / clip(dist,1,5)
// ---------------------------------------------------------------------------
__global__ __launch_bounds__(256) void build_attn_bias(
    const int* __restrict__ dist, const float* __restrict__ path_data,
    const float* __restrict__ spatial_tab, const float* __restrict__ path_tab,
    float* __restrict__ bias)
{
  size_t idx = (size_t)blockIdx.x * 256 + threadIdx.x;
  const size_t total = (size_t)B_ * N_ * N_;
  if (idx >= total) return;
  int d = dist[idx];
  int sidx = min(max(d, -1), MAXSPD) + 1;
  float rden = __builtin_amdgcn_rcpf((float)min(max(d, 1), MAXPATH));
  float pd[MAXPATH];
#pragma unroll
  for (int l = 0; l < MAXPATH; ++l) pd[l] = path_data[idx * MAXPATH + l];
  size_t b  = idx / ((size_t)N_ * N_);
  size_t xy = idx % ((size_t)N_ * N_);
#pragma unroll
  for (int h = 0; h < H_; ++h) {
    float pb = 0.f;
#pragma unroll
    for (int l = 0; l < MAXPATH; ++l) pb += pd[l] * path_tab[l * H_ + h];
    bias[(b * H_ + h) * (size_t)N_ * N_ + xy] = spatial_tab[sidx * H_ + h] + pb * rden;
  }
}

// ---------------------------------------------------------------------------
// Fused (optional residual) + LayerNorm (+ optional GELU), one row per block.
// ---------------------------------------------------------------------------
__global__ __launch_bounds__(256) void row_norm(
    const float* __restrict__ xin, const float* __restrict__ res,
    const float* __restrict__ g, const float* __restrict__ bta,
    float* __restrict__ out, int cols, int apply_gelu)
{
  const int row = blockIdx.x;
  const float* xr = xin + (size_t)row * cols;
  const float* rr = res ? res + (size_t)row * cols : nullptr;
  float* yr = out + (size_t)row * cols;
  const int nper = cols >> 8;
  float xs[4];
  float sum = 0.f, sq = 0.f;
  for (int t = 0; t < nper; ++t) {
    int c = threadIdx.x + (t << 8);
    float x = xr[c] + (rr ? rr[c] : 0.f);
    xs[t] = x; sum += x; sq += x * x;
  }
#pragma unroll
  for (int off = 16; off; off >>= 1) {
    sum += __shfl_xor(sum, off, 32);
    sq  += __shfl_xor(sq,  off, 32);
  }
  __shared__ float ssum[8], ssq[8];
  int wv = threadIdx.x >> 5;
  if ((threadIdx.x & 31) == 0) { ssum[wv] = sum; ssq[wv] = sq; }
  __syncthreads();
  sum = 0.f; sq = 0.f;
#pragma unroll
  for (int i = 0; i < 8; ++i) { sum += ssum[i]; sq += ssq[i]; }
  float inv_n = 1.0f / (float)cols;
  float mu  = sum * inv_n;
  float var = sq * inv_n - mu * mu;
  float rstd = rsqrtf(var + 1e-5f);
  for (int t = 0; t < nper; ++t) {
    int c = threadIdx.x + (t << 8);
    float y = (xs[t] - mu) * rstd * g[c] + bta[c];
    if (apply_gelu) y = gelu_exact(y);
    yr[c] = y;
  }
}

// ---------------------------------------------------------------------------
__global__ __launch_bounds__(256) void add_degree(
    float* __restrict__ h, const int* __restrict__ indeg,
    const int* __restrict__ outdeg,
    const float* __restrict__ tin, const float* __restrict__ tout)
{
  int bn = blockIdx.x;
  int c  = threadIdx.x;
  int di = min(max(indeg[bn], 0), MAXDEG);
  int dq = min(max(outdeg[bn], 0), MAXDEG);
  h[(size_t)bn * D_ + c] += tin[(size_t)di * D_ + c] + tout[(size_t)dq * D_ + c];
}

// ---------------------------------------------------------------------------
// Attention: one wave per (b, head, 16-query tile). q,k bf16 (B,N,D);
// vT bf16 (B,D,N).  HD=32 => one WMMA per 16-key score tile, with the Q
// A-fragment loaded once. Scores (16x1024 f32 = 64 KB) in LDS; wave32
// shuffle softmax; attn.V = two WMMAs (d-halves) per 32-key step with
// contiguous b128 B-fragment loads from vT. Output stored bf16.
// ---------------------------------------------------------------------------
__global__ __launch_bounds__(32) void attention_wmma(
    const __bf16* __restrict__ q, const __bf16* __restrict__ k,
    const __bf16* __restrict__ vT, const float* __restrict__ bias,
    __bf16* __restrict__ o)
{
  __shared__ float sc[16 * N_];   // 64 KB
  const int lane = threadIdx.x;
  const int hl  = lane >> 4;
  const int l16 = lane & 15;
  const int qt  = blockIdx.x << 4;
  const int hh  = blockIdx.y;
  const int bb  = blockIdx.z;
  const size_t base = (size_t)bb * N_ * D_ + (size_t)hh * HD_;
  const __bf16* qb = q + base;
  const __bf16* kb = k + base;
  // vT rows for this head: vT[b][hh*32 + d][*]
  const __bf16* vt = vT + (size_t)bb * N_ * D_ + (size_t)hh * HD_ * N_;

  v16bf aq = ldfrag(qb + (size_t)(qt + l16) * D_ + hl * 8);

  const float* bias_b = bias + ((size_t)(bb * H_ + hh) * N_ + qt) * N_;
  for (int kt = 0; kt < N_; kt += 16) {
    v16bf bk = ldfrag(kb + (size_t)(kt + l16) * D_ + hl * 8);
    v8f s = {0.f, 0.f, 0.f, 0.f, 0.f, 0.f, 0.f, 0.f};
    s = __builtin_amdgcn_wmma_f32_16x16x32_bf16(
        false, aq, false, bk, (short)0, s, false, false);
#pragma unroll
    for (int j = 0; j < 8; ++j) {
      int r = j + hl * 8;
      sc[r * N_ + kt + l16] = s[j] + bias_b[(size_t)r * N_ + kt + l16];
    }
  }
  __syncthreads();

  for (int r = 0; r < 16; ++r) {
    float* row = sc + r * N_;
    float m = -3.0e38f;
    for (int i = lane; i < N_; i += 32) m = fmaxf(m, row[i]);
#pragma unroll
    for (int off = 16; off; off >>= 1) m = fmaxf(m, __shfl_xor(m, off, 32));
    float s = 0.f;
    for (int i = lane; i < N_; i += 32) { float e = __expf(row[i] - m); row[i] = e; s += e; }
#pragma unroll
    for (int off = 16; off; off >>= 1) s += __shfl_xor(s, off, 32);
    float inv = __builtin_amdgcn_rcpf(s);
    for (int i = lane; i < N_; i += 32) row[i] *= inv;
  }
  __syncthreads();

  v8f o0 = {0.f, 0.f, 0.f, 0.f, 0.f, 0.f, 0.f, 0.f};
  v8f o1 = {0.f, 0.f, 0.f, 0.f, 0.f, 0.f, 0.f, 0.f};
  for (int kt = 0; kt < N_; kt += 32) {
    v16bf ap;
    const float* arow = sc + l16 * N_ + kt + hl * 8;
#pragma unroll
    for (int j = 0; j < 8; ++j) ap[j] = (__bf16)arow[j];
#pragma unroll
    for (int j = 0; j < 8; ++j) ap[8 + j] = (__bf16)arow[16 + j];
    v16bf b0 = ldfrag(vt + (size_t)l16 * N_ + kt + hl * 8);
    v16bf b1 = ldfrag(vt + (size_t)(16 + l16) * N_ + kt + hl * 8);
    o0 = __builtin_amdgcn_wmma_f32_16x16x32_bf16(
        false, ap, false, b0, (short)0, o0, false, false);
    o1 = __builtin_amdgcn_wmma_f32_16x16x32_bf16(
        false, ap, false, b1, (short)0, o1, false, false);
  }

  __bf16* op = o + base;
#pragma unroll
  for (int j = 0; j < 8; ++j) {
    int r = qt + j + hl * 8;
    op[(size_t)r * D_ + l16]      = (__bf16)o0[j];
    op[(size_t)r * D_ + 16 + l16] = (__bf16)o1[j];
  }
}

// ---------------------------------------------------------------------------
extern "C" void kernel_launch(void* const* d_in, const int* in_sizes, int n_in,
                              void* d_out, int out_size, void* d_ws, size_t ws_size,
                              hipStream_t stream)
{
  (void)in_sizes; (void)n_in; (void)out_size; (void)ws_size;

  const float* node_feat   = (const float*)d_in[0];
  const int*   in_degree   = (const int*)d_in[1];
  const int*   out_degree  = (const int*)d_in[2];
  const int*   dist        = (const int*)d_in[3];
  const float* path_data   = (const float*)d_in[4];
  const float* fc_w1       = (const float*)d_in[5];
  const float* fc_b1       = (const float*)d_in[6];
  const float* fc_ln_g     = (const float*)d_in[7];
  const float* fc_ln_b     = (const float*)d_in[8];
  const float* fc_w2       = (const float*)d_in[9];
  const float* fc_b2       = (const float*)d_in[10];
  const float* deg_in_tab  = (const float*)d_in[11];
  const float* deg_out_tab = (const float*)d_in[12];
  const float* spatial_tab = (const float*)d_in[13];
  const float* path_tab    = (const float*)d_in[14];
  const float* Wq = (const float*)d_in[15]; const float* bq = (const float*)d_in[16];
  const float* Wk = (const float*)d_in[17]; const float* bk = (const float*)d_in[18];
  const float* Wv = (const float*)d_in[19]; const float* bv = (const float*)d_in[20];
  const float* Wo = (const float*)d_in[21]; const float* bo = (const float*)d_in[22];
  const float* ln1_g = (const float*)d_in[23]; const float* ln1_b = (const float*)d_in[24];
  const float* ffn_w1 = (const float*)d_in[25]; const float* ffn_b1 = (const float*)d_in[26];
  const float* ffn_w2 = (const float*)d_in[27]; const float* ffn_b2 = (const float*)d_in[28];
  const float* ln2_g = (const float*)d_in[29]; const float* ln2_b = (const float*)d_in[30];
  const float* out_g = (const float*)d_in[31]; const float* out_b = (const float*)d_in[32];

  char* ws = (char*)d_ws;
  const size_t BN = (size_t)B_ * N_;   // 4096 rows
  size_t off = 0;
  float*  bias = (float*)(ws + off); off += (size_t)B_ * H_ * N_ * N_ * 4; // 134 MB
  float*  hbuf = (float*)(ws + off); off += BN * D_ * 4;        // 4 MB
  float*  t1   = (float*)(ws + off); off += BN * (2 * D_) * 4;  // 8 MB (fc inter)
  float*  t2   = (float*)(ws + off); off += BN * D_ * 4;        // 4 MB
  __bf16* nf_b = (__bf16*)(ws + off); off += (size_t)B_ * N_ * N_ * 2;  // 8 MB
  __bf16* h_b  = (__bf16*)(ws + off); off += BN * D_ * 2;       // 2 MB
  __bf16* t1b  = (__bf16*)(ws + off); off += BN * FF_ * 2;      // 8 MB
  __bf16* qb   = (__bf16*)(ws + off); off += BN * D_ * 2;
  __bf16* kbuf = (__bf16*)(ws + off); off += BN * D_ * 2;
  __bf16* vbuf = (__bf16*)(ws + off); off += BN * D_ * 2;
  __bf16* vTb  = (__bf16*)(ws + off); off += BN * D_ * 2;
  __bf16* obuf = (__bf16*)(ws + off); off += BN * D_ * 2;
  __bf16* wt   = (__bf16*)(ws + off); off += (size_t)1024 * 512 * 2;   // 1 MB scratch W^T

  {
    size_t total = (size_t)B_ * N_ * N_;
    build_attn_bias<<<dim3((unsigned)((total + 255) / 256)), dim3(256), 0, stream>>>(
        dist, path_data, spatial_tab, path_tab, bias);
  }

  // --- feature MLP: Linear(1024,512) -> LN -> GELU -> Linear(512,256) ---
  conv_bf16<<<dim3((unsigned)(B_ * N_ * N_ / 1024)), dim3(256), 0, stream>>>(
      node_feat, nf_b, B_ * N_ * N_);
  transpose_conv_bf16<<<dim3((2 * D_) / 32, N_ / 32), dim3(256), 0, stream>>>(
      fc_w1, wt, N_, 2 * D_);
  gemm_bf16_wmma<<<dim3((2 * D_) / 64, (unsigned)(BN / 16)), dim3(32), 0, stream>>>(
      nf_b, wt, fc_b1, t1, nullptr, (int)BN, N_, 2 * D_, 1.0f, 0);
  row_norm<<<dim3((unsigned)BN), dim3(256), 0, stream>>>(
      t1, nullptr, fc_ln_g, fc_ln_b, t1, 2 * D_, 1);
  conv_bf16<<<dim3((unsigned)(BN * 2 * D_ / 1024)), dim3(256), 0, stream>>>(
      t1, t1b, (int)(BN * 2 * D_));
  transpose_conv_bf16<<<dim3(D_ / 32, (2 * D_) / 32), dim3(256), 0, stream>>>(
      fc_w2, wt, 2 * D_, D_);
  gemm_bf16_wmma<<<dim3(D_ / 64, (unsigned)(BN / 16)), dim3(32), 0, stream>>>(
      t1b, wt, fc_b2, hbuf, nullptr, (int)BN, 2 * D_, D_, 1.0f, 0);
  add_degree<<<dim3((unsigned)BN), dim3(256), 0, stream>>>(
      hbuf, in_degree, out_degree, deg_in_tab, deg_out_tab);

  const float scale = 0.17677669529663689f;  // HD^-0.5
  for (int i = 0; i < L_; ++i) {
    conv_bf16<<<dim3((unsigned)(BN * D_ / 1024)), dim3(256), 0, stream>>>(
        hbuf, h_b, (int)(BN * D_));
    // Q
    transpose_conv_bf16<<<dim3(D_ / 32, D_ / 32), dim3(256), 0, stream>>>(
        Wq + (size_t)i * D_ * D_, wt, D_, D_);
    gemm_bf16_wmma<<<dim3(D_ / 64, (unsigned)(BN / 16)), dim3(32), 0, stream>>>(
        h_b, wt, bq + (size_t)i * D_, nullptr, qb, (int)BN, D_, D_, scale, 0);
    // K
    transpose_conv_bf16<<<dim3(D_ / 32, D_ / 32), dim3(256), 0, stream>>>(
        Wk + (size_t)i * D_ * D_, wt, D_, D_);
    gemm_bf16_wmma<<<dim3(D_ / 64, (unsigned)(BN / 16)), dim3(32), 0, stream>>>(
        h_b, wt, bk + (size_t)i * D_, nullptr, kbuf, (int)BN, D_, D_, 1.0f, 0);
    // V (+ transpose to (B,D,N))
    transpose_conv_bf16<<<dim3(D_ / 32, D_ / 32), dim3(256), 0, stream>>>(
        Wv + (size_t)i * D_ * D_, wt, D_, D_);
    gemm_bf16_wmma<<<dim3(D_ / 64, (unsigned)(BN / 16)), dim3(32), 0, stream>>>(
        h_b, wt, bv + (size_t)i * D_, nullptr, vbuf, (int)BN, D_, D_, 1.0f, 0);
    transpose_v_bf16<<<dim3(D_ / 32, N_ / 32, B_), dim3(256), 0, stream>>>(vbuf, vTb);
    // attention
    attention_wmma<<<dim3(N_ / 16, H_, B_), dim3(32), 0, stream>>>(
        qb, kbuf, vTb, bias, obuf);
    // O projection + residual LN
    transpose_conv_bf16<<<dim3(D_ / 32, D_ / 32), dim3(256), 0, stream>>>(
        Wo + (size_t)i * D_ * D_, wt, D_, D_);
    gemm_bf16_wmma<<<dim3(D_ / 64, (unsigned)(BN / 16)), dim3(32), 0, stream>>>(
        obuf, wt, bo + (size_t)i * D_, t2, nullptr, (int)BN, D_, D_, 1.0f, 0);
    row_norm<<<dim3((unsigned)BN), dim3(256), 0, stream>>>(
        hbuf, t2, ln1_g + (size_t)i * D_, ln1_b + (size_t)i * D_, hbuf, D_, 0);
    // FFN
    conv_bf16<<<dim3((unsigned)(BN * D_ / 1024)), dim3(256), 0, stream>>>(
        hbuf, h_b, (int)(BN * D_));
    transpose_conv_bf16<<<dim3(FF_ / 32, D_ / 32), dim3(256), 0, stream>>>(
        ffn_w1 + (size_t)i * D_ * FF_, wt, D_, FF_);
    gemm_bf16_wmma<<<dim3(FF_ / 64, (unsigned)(BN / 16)), dim3(32), 0, stream>>>(
        h_b, wt, ffn_b1 + (size_t)i * FF_, nullptr, t1b, (int)BN, D_, FF_, 1.0f, 1);
    transpose_conv_bf16<<<dim3(D_ / 32, FF_ / 32), dim3(256), 0, stream>>>(
        ffn_w2 + (size_t)i * FF_ * D_, wt, FF_, D_);
    gemm_bf16_wmma<<<dim3(D_ / 64, (unsigned)(BN / 16)), dim3(32), 0, stream>>>(
        t1b, wt, ffn_b2 + (size_t)i * D_, t2, nullptr, (int)BN, FF_, D_, 1.0f, 0);
    row_norm<<<dim3((unsigned)BN), dim3(256), 0, stream>>>(
        hbuf, t2, ln2_g + (size_t)i * D_, ln2_b + (size_t)i * D_, hbuf, D_, 0);
  }

  row_norm<<<dim3((unsigned)BN), dim3(256), 0, stream>>>(
      hbuf, nullptr, out_g, out_b, (float*)d_out, D_, 0);
}
